// RPN_Modulator_28518582845486
// MI455X (gfx1250) — matrix operand order repaired and checked by reading, not compiled, not run on hardware
//
#include <hip/hip_runtime.h>

// ---------------------------------------------------------------------------
// MI455X / gfx1250 implementation of batched RPN_Modulator inference.
// All matrix math lowered to v_wmma_f32_16x16x32_f16 (wave32).
//
// Sizes: B=4, I=4, C=256, K=7, L=5, HW = {160^2, 80^2, 40^2, 20^2, 10^2}.
// Workspace layout (bytes):
//   [0,            81920)      q      f32 [5][16][256]
//   [81920,        10567680)   W2     f16 [5][16][256][256]  (po_w * q)
//   [10567680,     80404480)   featsT f16 [5][4][HW][256]    (transposed, f16)
// ---------------------------------------------------------------------------

typedef _Float16 v16h __attribute__((ext_vector_type(16)));
typedef _Float16 v8h  __attribute__((ext_vector_type(8)));
typedef float    v8f  __attribute__((ext_vector_type(8)));

union F16Frag { v16h v; v8h h[2]; _Float16 e[16]; };

#define CKK 12544   // 256*7*7 contraction length for the q GEMM (= 392 * 32)

// ---------------------------------------------------------------------------
// Stage 1: q[k][bi][o] = sum_kk modulator[bi][kk] * pm_w[k][o][kk] + pm_b[k][o]
// One wave per (k, o-tile). M=16 (bi) fits one WMMA tile exactly; K=12544 is
// an exact multiple of 32. Both operands are contiguous per lane, converted
// f32->f16 on the fly; accumulation stays f32 in the WMMA accumulator.
// ---------------------------------------------------------------------------
__global__ __launch_bounds__(32) void q_gemm_kernel(
    const float* __restrict__ modulator,  // [16][CKK]
    const float* __restrict__ pm_w,       // [5][256][CKK]
    const float* __restrict__ pm_b,       // [5][256]
    float* __restrict__ qout)             // [5][16][256]
{
    const int lane    = threadIdx.x;
    const int k       = blockIdx.x >> 4;
    const int o0      = (blockIdx.x & 15) << 4;
    const int half_hi = lane >> 4;
    const int mn      = lane & 15;
    const int koffA   = half_hi ? 16 : 0;   // A 16x32 layout: K split at 16
    const int kbaseB  = half_hi ? 16 : 0;   // B 32x16 layout: K split at 16

    const float* Arow = modulator + (size_t)mn * CKK;
    const float* Brow = pm_w + ((size_t)k * 256 + (size_t)(o0 + mn)) * CKK;

    v8f acc = {};
    for (int c0 = 0; c0 < CKK; c0 += 32) {
        F16Frag a, b;
        const float* pa0 = Arow + c0 + koffA;
        const float* pa1 = Arow + c0 + 8 + koffA;
#pragma unroll
        for (int t = 0; t < 8; ++t) {
            a.e[t]     = (_Float16)pa0[t];
            a.e[8 + t] = (_Float16)pa1[t];
        }
        const float* pb = Brow + c0 + kbaseB;
#pragma unroll
        for (int t = 0; t < 16; ++t) b.e[t] = (_Float16)pb[t];

        acc = __builtin_amdgcn_wmma_f32_16x16x32_f16(
            false, a.v, false, b.v, (short)0, acc, false, false);
    }

    const float bias = pm_b[k * 256 + o0 + mn];
#pragma unroll
    for (int r = 0; r < 8; ++r) {
        const int bi = r + half_hi * 8;                 // D: M = r + 8*(lane>=16)
        qout[((size_t)k * 16 + bi) * 256 + (o0 + mn)] = acc[r] + bias;
    }
}

// ---------------------------------------------------------------------------
// Stage 2: W2[k][bi][o][c] = f16( po_w[k][o][c] * q[k][bi][c] )
// Folds the channel-wise modulation into the A operand of the big GEMM.
// 5*16*256*256 = 5,242,880 elements, 4 per thread -> 5120 blocks exactly.
// ---------------------------------------------------------------------------
__global__ __launch_bounds__(256) void w2_build_kernel(
    const float* __restrict__ po_w,   // [5][256][256]
    const float* __restrict__ q,      // [5][16][256]
    _Float16* __restrict__ w2)        // [5][16][256][256]
{
    const size_t e = ((size_t)blockIdx.x * 256 + threadIdx.x) * 4;
    const int c  = (int)(e & 255);
    const int o  = (int)((e >> 8) & 255);
    const int bi = (int)((e >> 16) & 15);
    const int k  = (int)(e >> 20);

    const float4 pw = *(const float4*)(po_w + ((size_t)k * 256 + o) * 256 + c);
    const float4 qq = *(const float4*)(q + ((size_t)k * 16 + bi) * 256 + c);

    union { _Float16 h[4]; unsigned long long u; } r;
    r.h[0] = (_Float16)(pw.x * qq.x);
    r.h[1] = (_Float16)(pw.y * qq.y);
    r.h[2] = (_Float16)(pw.z * qq.z);
    r.h[3] = (_Float16)(pw.w * qq.w);
    *(unsigned long long*)(w2 + e) = r.u;
}

// ---------------------------------------------------------------------------
// Stage 3: featsT[b][hw][c] = f16(feats[b][c][hw])  (LDS-tiled transpose).
// Makes the WMMA B-fragment a contiguous per-lane run of 16 halves.
// ---------------------------------------------------------------------------
__global__ __launch_bounds__(256) void transpose_kernel(
    const float* __restrict__ feats,  // [4][256][HW]
    _Float16* __restrict__ out,       // [4][HW][256]
    int HW)
{
    __shared__ _Float16 tile[32][33];
    const int hw0 = blockIdx.x * 32;
    const int c0  = blockIdx.y * 32;
    const int b   = blockIdx.z;
    const int tx  = threadIdx.x;
    const int ty0 = threadIdx.y;

    const float* in  = feats + (size_t)b * 256 * HW;
    _Float16*    po  = out   + (size_t)b * HW * 256;

#pragma unroll
    for (int j = 0; j < 4; ++j) {
        const int ty = ty0 + j * 8;
        const int hw = hw0 + tx;
        const int c  = c0 + ty;
        tile[ty][tx] = (hw < HW) ? (_Float16)in[(size_t)c * HW + hw] : (_Float16)0.0f;
    }
    __syncthreads();
#pragma unroll
    for (int j = 0; j < 4; ++j) {
        const int r  = ty0 + j * 8;
        const int hw = hw0 + r;
        const int c  = c0 + tx;
        if (hw < HW) po[(size_t)hw * 256 + c] = tile[tx][r];
    }
}

// ---------------------------------------------------------------------------
// Stage 4 (main GEMM, per level): out[bi][o][hw] = sum_c W2[bi][o][c] *
// featsT[b][hw][c] + po_b[o].  Each wave owns a 16(M)x64(N) strip:
// 4 f32 accumulators, K-loop of 8 -> 32 v_wmma per wave.  A fragment = two
// b128 loads from W2 (one contiguous 16-half K run per lane); B fragment = two b128
// loads from featsT.  Out-of-range columns get zero B data / masked stores so
// EXEC stays all-1s across WMMA.
// ---------------------------------------------------------------------------
__global__ __launch_bounds__(256) void modulate_gemm_kernel(
    const _Float16* __restrict__ featsT,  // [4][HW][256] (this level)
    const _Float16* __restrict__ w2,      // [16][256][256] (this level)
    const float* __restrict__ po_b_k,     // [256]
    float* __restrict__ out,              // [16][256][HW] (this level)
    int HW, int NC, int jobs)
{
    const int wid = blockIdx.x * 8 + (threadIdx.x >> 5);
    if (wid >= jobs) return;                 // uniform per wave

    const int lane    = threadIdx.x & 31;
    const int nc      = wid % NC;
    const int rest    = wid / NC;
    const int mt      = rest & 15;
    const int bi      = rest >> 4;
    const int b       = bi >> 2;
    const int o0      = mt << 4;
    const int hw0     = nc << 6;
    const int half_hi = lane >> 4;
    const int mn      = lane & 15;
    const int koffA   = half_hi ? 16 : 0;
    const int kbaseB  = half_hi ? 16 : 0;

    const _Float16* w2row = w2 + ((size_t)bi * 256 + (size_t)(o0 + mn)) * 256;
    const _Float16* ftb   = featsT + (size_t)b * HW * 256;

    const F16Frag zero = {};
    v8f acc[4] = {{}, {}, {}, {}};

    for (int c0 = 0; c0 < 256; c0 += 32) {
        F16Frag a;
        a.h[0] = *(const v8h*)(w2row + c0 + koffA);
        a.h[1] = *(const v8h*)(w2row + c0 + 8 + koffA);
#pragma unroll
        for (int j = 0; j < 4; ++j) {
            const int n = hw0 + j * 16 + mn;
            F16Frag bf;
            if (n < HW) {
                const _Float16* pb = ftb + (size_t)n * 256 + c0 + kbaseB;
                bf.h[0] = *(const v8h*)(pb);
                bf.h[1] = *(const v8h*)(pb + 8);
            } else {
                bf = zero;
            }
            acc[j] = __builtin_amdgcn_wmma_f32_16x16x32_f16(
                false, a.v, false, bf.v, (short)0, acc[j], false, false);
        }
    }

#pragma unroll
    for (int j = 0; j < 4; ++j) {
        const int n = hw0 + j * 16 + mn;
        if (n < HW) {
#pragma unroll
            for (int r = 0; r < 8; ++r) {
                const int o = o0 + r + half_hi * 8;
                float v = acc[j][r] + po_b_k[o];
                __builtin_nontemporal_store(
                    v, out + ((size_t)bi * 256 + o) * HW + n);
            }
        }
    }
}

// ---------------------------------------------------------------------------
extern "C" void kernel_launch(void* const* d_in, const int* in_sizes, int n_in,
                              void* d_out, int out_size, void* d_ws, size_t ws_size,
                              hipStream_t stream) {
    (void)in_sizes; (void)n_in; (void)out_size; (void)ws_size;

    const float* feats[5];
    for (int k = 0; k < 5; ++k) feats[k] = (const float*)d_in[k];
    const float* modulator = (const float*)d_in[5];
    const float* pm_w      = (const float*)d_in[6];
    const float* pm_b      = (const float*)d_in[7];
    const float* po_w      = (const float*)d_in[8];
    const float* po_b      = (const float*)d_in[9];
    float* out = (float*)d_out;

    char* ws = (char*)d_ws;
    float*     q_buf  = (float*)ws;                               // 81,920 B
    _Float16*  w2_buf = (_Float16*)(ws + 81920);                  // 10,485,760 B
    _Float16*  ft_buf = (_Float16*)(ws + 81920 + 10485760);       // 69,836,800 B

    static const int HWs[5] = {25600, 6400, 1600, 400, 100};

    size_t ft_off[5], out_off[5];
    {
        size_t fa = 0, oa = 0;
        for (int k = 0; k < 5; ++k) {
            ft_off[k]  = fa;  fa += (size_t)4 * HWs[k] * 256;      // halves
            out_off[k] = oa;  oa += (size_t)16 * 256 * HWs[k];     // floats
        }
    }

    // Stage 1: q (80 waves)
    q_gemm_kernel<<<80, 32, 0, stream>>>(modulator, pm_w, pm_b, q_buf);

    // Stage 2: W2 = po_w * q (f16)
    w2_build_kernel<<<5120, 256, 0, stream>>>(po_w, q_buf, w2_buf);

    // Stage 3: feats -> featsT (f16), per level
    for (int k = 0; k < 5; ++k) {
        dim3 grid((HWs[k] + 31) / 32, 8, 4);
        transpose_kernel<<<grid, dim3(32, 8), 0, stream>>>(
            feats[k], ft_buf + ft_off[k], HWs[k]);
    }

    // Stage 4: main WMMA GEMM, per level
    for (int k = 0; k < 5; ++k) {
        const int NC    = (HWs[k] + 63) / 64;
        const int jobs  = 256 * NC;            // 16 bi * 16 m-tiles * NC
        const int blocks = (jobs + 7) / 8;
        modulate_gemm_kernel<<<blocks, 256, 0, stream>>>(
            ft_buf + ft_off[k],
            w2_buf + (size_t)k * 16 * 256 * 256,
            po_b + (size_t)k * 256,
            out + out_off[k],
            HWs[k], NC, jobs);
    }
}
